// SepConvVisualHead_26628797235436
// MI455X (gfx1250) — matrix operand order repaired and checked by reading, not compiled, not run on hardware
//
#include <hip/hip_runtime.h>
#include <hip/hip_bf16.h>
#include <stdint.h>

// ---------------------------------------------------------------------------
// Problem constants (match reference)
// ---------------------------------------------------------------------------
#define BB    2048
#define CC    1024
#define VV    8000
#define VPAD  8064          // 63 * 128  (N / M padding for guard-free tiles)
#define EE    300
#define EPAD  320           // K padding to multiple of 32
#define KK    5
#define BK    (BB * KK)     // 10240

typedef __attribute__((ext_vector_type(16))) __bf16 v16bf;
typedef __attribute__((ext_vector_type(8)))  float  v8f;

// fp32 -> bf16 round-to-nearest-even (bit trick), stored as raw u16
__device__ __forceinline__ unsigned short f2bf(float f) {
    unsigned u = __float_as_uint(f);
    u += 0x7FFFu + ((u >> 16) & 1u);
    return (unsigned short)(u >> 16);
}

// ---------------------------------------------------------------------------
// CDNA5 async global->LDS copy (ASYNCcnt path, no VGPR round trip).
// LDS operand is the wave-relative byte offset == low 32 bits of the flat
// shared-aperture address (ISA 10.2 aperture mapping).
// ---------------------------------------------------------------------------
__device__ __forceinline__ void async_load_b128(const void* gptr, void* lptr) {
    unsigned           loff = (unsigned)(uintptr_t)lptr;
    unsigned long long ga   = (unsigned long long)(uintptr_t)gptr;
    asm volatile("global_load_async_to_lds_b128 %0, %1, off"
                 :: "v"(loff), "v"(ga) : "memory");
}
__device__ __forceinline__ void wait_async0() {
    asm volatile("s_wait_asynccnt 0x0" ::: "memory");
}

// ---------------------------------------------------------------------------
// Conversion kernels
// ---------------------------------------------------------------------------
__global__ void cvt_bf16_kernel(const float* __restrict__ src,
                                unsigned short* __restrict__ dst, long n) {
    long i = (long)blockIdx.x * blockDim.x + threadIdx.x;
    if (i < n) dst[i] = f2bf(src[i]);
}

// src [R][C] f32 -> dst [RPad][CPad] bf16, zero padded
__global__ void cvt_pad2_kernel(const float* __restrict__ src,
                                unsigned short* __restrict__ dst,
                                int R, int C_, int RPad, int CPad) {
    long i = (long)blockIdx.x * blockDim.x + threadIdx.x;
    long n = (long)RPad * CPad;
    if (i >= n) return;
    int  c = (int)(i % CPad);
    long r = i / CPad;
    dst[i] = (r < R && c < C_) ? f2bf(src[r * C_ + c]) : (unsigned short)0;
}

// Tiled transpose + convert: src [R][C] f32 -> dst [CPad][RPad] bf16
// dst[c][r] = bf16(src[r][c]); zero padding outside. CPad/RPad multiples of 32.
// grid = (CPad/32, RPad/32), block = (32, 8)
__global__ __launch_bounds__(256)
void transpose_cvt_kernel(const float* __restrict__ src,
                          unsigned short* __restrict__ dst,
                          int R, int C_, int CPad, int RPad) {
    __shared__ float t[32][33];
    const int tx = threadIdx.x;
    const int ty = threadIdx.y;
    const int bx = blockIdx.x;   // tiles over dst rows (= src cols)
    const int by = blockIdx.y;   // tiles over dst cols (= src rows)

    #pragma unroll
    for (int j = 0; j < 4; ++j) {
        int r = by * 32 + ty + j * 8;   // src row
        int c = bx * 32 + tx;           // src col
        t[ty + j * 8][tx] = (r < R && c < C_) ? src[(long)r * C_ + c] : 0.0f;
    }
    __syncthreads();
    #pragma unroll
    for (int j = 0; j < 4; ++j) {
        int dr = bx * 32 + ty + j * 8;  // dst row (src col)
        int dc = by * 32 + tx;          // dst col (src row)
        dst[(long)dr * RPad + dc] = f2bf(t[tx][ty + j * 8]);
    }
}

// ---------------------------------------------------------------------------
// WMMA bf16 GEMM:  C[M,N] = A[M,Kd](bf16) * Bt[N,Kd](bf16, pre-transposed)
//                            + bias[N]   (fp32 accumulate / fp32 out)
// A padded to >= gridDim.y*128 rows, Bt padded to >= gridDim.x*128 rows,
// Kd multiple of 32  ==>  interior loads are guard-free.
// Block: 256 threads = 8 waves (wave32). Block tile 128x128.
// Waves 4(M) x 2(N); wave tile 32x64 = 2x4 WMMA 16x16x32 accumulators.
// Double-buffered LDS, next tile staged via async global->LDS (ASYNCcnt).
// ---------------------------------------------------------------------------
__global__ __launch_bounds__(256)
void gemm_bf16_wmma_kernel(const unsigned short* __restrict__ A,
                           const unsigned short* __restrict__ Bt,
                           const float* __restrict__ bias,
                           float* __restrict__ Cout,
                           int M, int N, int Kd) {
    __shared__ __attribute__((aligned(16))) unsigned short ldsA[2][128][32]; // [buf][m][k]
    __shared__ __attribute__((aligned(16))) unsigned short ldsB[2][128][32]; // [buf][n][k]

    const int tid  = threadIdx.x;
    const int lane = tid & 31;
    const int wave = tid >> 5;
    const int waveM = wave >> 1;   // 0..3
    const int waveN = wave & 1;    // 0..1
    const int tileM = blockIdx.y * 128;
    const int tileN = blockIdx.x * 128;

    // staging: tile = 128 rows x 32 bf16 = 512 x 16B chunks; this thread
    // moves chunks {tid, tid+256} of both A and Bt tiles
    const int row0 = tid >> 2;               // 0..63
    const int col0 = (tid & 3) * 8;          // 0,8,16,24
    const int row1 = row0 + 64;              // 64..127

    const unsigned short* pA0 = A  + (long)(tileM + row0) * Kd + col0;
    const unsigned short* pA1 = A  + (long)(tileM + row1) * Kd + col0;
    const unsigned short* pB0 = Bt + (long)(tileN + row0) * Kd + col0;
    const unsigned short* pB1 = Bt + (long)(tileN + row1) * Kd + col0;

    v8f acc[2][4] = {};
    union Frag { v16bf v; uint4 q[2]; };

    const int m16    = lane & 15;
    const int laneHi = lane >> 4;           // 0 or 1
    const int khalfA = laneHi ? 8 : 0;      // A frag: K pieces per ISA layout
    const int khalfB = laneHi * 8;          // B frag: K pieces per ISA layout

    // ---- prologue: async-stage tile 0 into buffer 0 ----
    async_load_b128(pA0, &ldsA[0][row0][col0]);
    async_load_b128(pA1, &ldsA[0][row1][col0]);
    async_load_b128(pB0, &ldsB[0][row0][col0]);
    async_load_b128(pB1, &ldsB[0][row1][col0]);
    wait_async0();
    __syncthreads();

    int cur = 0;
    for (int kk = 0; kk < Kd; kk += 32, cur ^= 1) {
        const bool more = (kk + 32) < Kd;
        const int  nxt  = cur ^ 1;

        // ---- async-stage tile k+1 into the other buffer (overlaps WMMAs) ----
        if (more) {
            async_load_b128(pA0 + kk + 32, &ldsA[nxt][row0][col0]);
            async_load_b128(pA1 + kk + 32, &ldsA[nxt][row1][col0]);
            async_load_b128(pB0 + kk + 32, &ldsB[nxt][row0][col0]);
            async_load_b128(pB1 + kk + 32, &ldsB[nxt][row1][col0]);
        }
        // prefetch tile k+2 into caches
        if (kk + 64 < Kd) {
            if (tid < 128)
                __builtin_prefetch(A  + (long)(tileM + tid) * Kd + kk + 64, 0, 1);
            else
                __builtin_prefetch(Bt + (long)(tileN + tid - 128) * Kd + kk + 64, 0, 1);
        }

        // ---- fragments from LDS (contiguous 2x b128 per lane) ----
        Frag af[2], bf[4];
        #pragma unroll
        for (int mi = 0; mi < 2; ++mi) {
            int arow = waveM * 32 + mi * 16 + m16;
            af[mi].q[0] = *(const uint4*)(&ldsA[cur][arow][khalfA]);
            af[mi].q[1] = *(const uint4*)(&ldsA[cur][arow][khalfA + 16]);
        }
        #pragma unroll
        for (int ni = 0; ni < 4; ++ni) {
            int brow = waveN * 64 + ni * 16 + m16;
            bf[ni].q[0] = *(const uint4*)(&ldsB[cur][brow][khalfB]);
            bf[ni].q[1] = *(const uint4*)(&ldsB[cur][brow][khalfB + 16]);
        }

        // ---- 8x V_WMMA_F32_16X16X32_BF16 ----
        #pragma unroll
        for (int mi = 0; mi < 2; ++mi)
            #pragma unroll
            for (int ni = 0; ni < 4; ++ni)
                acc[mi][ni] = __builtin_amdgcn_wmma_f32_16x16x32_bf16(
                    false, af[mi].v, false, bf[ni].v,
                    (short)0, acc[mi][ni], false, false);

        wait_async0();      // our async writes to buf[nxt] have landed
        __syncthreads();    // everyone's reads of buf[cur] + writes of buf[nxt] done
    }

    // ---- epilogue: C/D layout -> global, + bias ----
    const int rowHalf = laneHi * 8;
    const int colBase = tileN + waveN * 64;
    const bool interior = (tileM + 128 <= M) && (tileN + 128 <= N);

    if (interior) {
        // branch-free bulk stores
        #pragma unroll
        for (int mi = 0; mi < 2; ++mi) {
            #pragma unroll
            for (int ni = 0; ni < 4; ++ni) {
                int   col = colBase + ni * 16 + m16;
                float bv  = bias[col];
                long  base = (long)(tileM + waveM * 32 + mi * 16 + rowHalf) * N + col;
                #pragma unroll
                for (int r = 0; r < 8; ++r)
                    Cout[base + (long)r * N] = acc[mi][ni][r] + bv;
            }
        }
    } else {
        #pragma unroll
        for (int mi = 0; mi < 2; ++mi) {
            #pragma unroll
            for (int ni = 0; ni < 4; ++ni) {
                int col = colBase + ni * 16 + m16;
                if (col < N) {
                    float bv = bias[col];
                    #pragma unroll
                    for (int r = 0; r < 8; ++r) {
                        int row = tileM + waveM * 32 + mi * 16 + rowHalf + r;
                        if (row < M)
                            Cout[(long)row * N + col] = acc[mi][ni][r] + bv;
                    }
                }
            }
        }
    }
}

// ---------------------------------------------------------------------------
// Top-5 per row (one 256-thread block per row). Tie-break: lower index wins.
// ---------------------------------------------------------------------------
__global__ __launch_bounds__(256)
void topk_kernel(const float* __restrict__ logits, int* __restrict__ outIdx) {
    const int row = blockIdx.x;
    const int tid = threadIdx.x;
    const float* lp = logits + (long)row * VV;

    float vals[32];
    int cnt = 0;
    for (int j = tid; j < VV; j += 256) vals[cnt++] = lp[j];

    __shared__ float sval[256];
    __shared__ int   sidx[256];

    for (int r = 0; r < KK; ++r) {
        float best = -3.0e38f;
        int   bj   = 0x7FFFFFFF;
        for (int t = 0; t < cnt; ++t) {
            int j = tid + t * 256;
            if (vals[t] > best || (vals[t] == best && j < bj)) {
                best = vals[t]; bj = j;
            }
        }
        sval[tid] = best;
        sidx[tid] = bj;
        __syncthreads();
        for (int s = 128; s > 0; s >>= 1) {
            if (tid < s) {
                float ov = sval[tid + s];
                int   oi = sidx[tid + s];
                if (ov > sval[tid] || (ov == sval[tid] && oi < sidx[tid])) {
                    sval[tid] = ov; sidx[tid] = oi;
                }
            }
            __syncthreads();
        }
        int win = sidx[0];
        if (tid == 0) outIdx[row * KK + r] = win;
        if ((win & 255) == tid) vals[win >> 8] = -3.0e38f;  // mask out winner
        __syncthreads();
    }
}

// ---------------------------------------------------------------------------
// fused_bf16[(b*5+k)][c] = bf16( x[b][c] + k_tab[idx[b*5+k]][c] )
// ---------------------------------------------------------------------------
__global__ void fuse_kernel(const float* __restrict__ x,
                            const float* __restrict__ ktab,
                            const int* __restrict__ idx,
                            unsigned short* __restrict__ fused) {
    long i = (long)blockIdx.x * blockDim.x + threadIdx.x;
    long n = (long)BK * CC;
    if (i >= n) return;
    int  c  = (int)(i % CC);
    long bk = i / CC;
    long b  = bk / KK;
    int  w  = idx[bk];
    fused[i] = f2bf(x[b * CC + c] + ktab[(long)w * CC + c]);
}

// ---------------------------------------------------------------------------
// Launch
// ---------------------------------------------------------------------------
extern "C" void kernel_launch(void* const* d_in, const int* in_sizes, int n_in,
                              void* d_out, int out_size, void* d_ws, size_t ws_size,
                              hipStream_t stream) {
    const float* x       = (const float*)d_in[0];  // [B,C]
    const float* W_out   = (const float*)d_in[1];  // [C,V]
    const float* b_out   = (const float*)d_in[2];  // [V]
    const float* emb     = (const float*)d_in[3];  // [V,E]
    const float* W_map   = (const float*)d_in[4];  // [E,C]
    const float* b_map   = (const float*)d_in[5];  // [C]
    const float* W_fused = (const float*)d_in[6];  // [C,V]
    const float* b_fused = (const float*)d_in[7];  // [V]

    float* out_logits = (float*)d_out;                      // B*V
    float* out_fused  = out_logits + (long)BB * VV;         // B*K*V
    int*   out_idx    = (int*)(out_fused + (long)BK * VV);  // B*K

    // workspace carve-up (256B aligned slabs)
    char* p = (char*)d_ws;
    auto take = [&](size_t bytes) {
        char* r = p;
        p += (bytes + 255) & ~(size_t)255;
        return r;
    };
    unsigned short* xb    = (unsigned short*)take((size_t)BB * CC * 2);      // [2048][1024]
    unsigned short* wobT  = (unsigned short*)take((size_t)VPAD * CC * 2);    // [8064][1024]  = W_out^T
    unsigned short* wfbT  = (unsigned short*)take((size_t)VPAD * CC * 2);    // [8064][1024]  = W_fused^T
    unsigned short* embb  = (unsigned short*)take((size_t)VPAD * EPAD * 2);  // [8064][320]
    unsigned short* wmapT = (unsigned short*)take((size_t)CC * EPAD * 2);    // [1024][320]   = W_map^T
    float*          ktab  = (float*)take((size_t)VV * CC * 4);               // [8000][1024]
    unsigned short* fusb  = (unsigned short*)take((size_t)BK * CC * 2);      // [10240][1024]

    const int T = 256;
    auto gsz = [](long n, int t) { return (unsigned)((n + t - 1) / t); };

    // 1) conversions: bf16, pre-transposed weights, tile-padded
    cvt_bf16_kernel<<<gsz((long)BB * CC, T), T, 0, stream>>>(x, xb, (long)BB * CC);
    {   // W_out [1024][8000] -> wobT [8064][1024]
        dim3 grid(VPAD / 32, CC / 32), blk(32, 8);
        transpose_cvt_kernel<<<grid, blk, 0, stream>>>(W_out, wobT, CC, VV, VPAD, CC);
    }
    {   // W_fused [1024][8000] -> wfbT [8064][1024]
        dim3 grid(VPAD / 32, CC / 32), blk(32, 8);
        transpose_cvt_kernel<<<grid, blk, 0, stream>>>(W_fused, wfbT, CC, VV, VPAD, CC);
    }
    {   // W_map [300][1024] -> wmapT [1024][320]
        dim3 grid(CC / 32, EPAD / 32), blk(32, 8);
        transpose_cvt_kernel<<<grid, blk, 0, stream>>>(W_map, wmapT, EE, CC, CC, EPAD);
    }
    // emb [8000][300] -> embb [8064][320]
    cvt_pad2_kernel<<<gsz((long)VPAD * EPAD, T), T, 0, stream>>>(emb, embb, VV, EE, VPAD, EPAD);

    // 2) logits = x @ W_out + b_out   [2048 x 8000], Kd=1024
    {
        dim3 grid(VPAD / 128, BB / 128);
        gemm_bf16_wmma_kernel<<<grid, T, 0, stream>>>(xb, wobT, b_out, out_logits,
                                                      BB, VV, CC);
    }

    // 3) k_tab = emb @ W_map + b_map  [8000 x 1024], Kd=320
    {
        dim3 grid(CC / 128, VPAD / 128);
        gemm_bf16_wmma_kernel<<<grid, T, 0, stream>>>(embb, wmapT, b_map, ktab,
                                                      VV, CC, EPAD);
    }

    // 4) top-5 indices per row
    topk_kernel<<<BB, T, 0, stream>>>(out_logits, out_idx);

    // 5) fused_fea (bf16) = x + gather(k_tab, idx)
    fuse_kernel<<<gsz((long)BK * CC, T), T, 0, stream>>>(x, ktab, out_idx, fusb);

    // 6) word_fused_gloss_logits = fused @ W_fused + b_fused  [10240 x 8000]
    {
        dim3 grid(VPAD / 128, BK / 128);
        gemm_bf16_wmma_kernel<<<grid, T, 0, stream>>>(fusb, wfbT, b_fused, out_fused,
                                                      BK, VV, CC);
    }
}
